// unit_san_6811818131974
// MI455X (gfx1250) — compile-verified, matrix operand order checked by reading, not run on hardware
//
#include <hip/hip_runtime.h>
#include <hip/hip_bf16.h>
#include <math.h>

typedef __attribute__((ext_vector_type(16))) _Float16 v16h;
typedef __attribute__((ext_vector_type(8)))  float    v8f;

#define NB  64
#define TT  64
#define VV  27
#define CINN 64
#define CC  256
#define KH  8
#define CIH 32
#define HH  1024
#define EE  6
#define MM  (NB*TT*VV)   /* 110592 = 32 * 3456 */

// ---------------------------------------------------------------------------
// wave32 reduction
// ---------------------------------------------------------------------------
__device__ __forceinline__ float wsum32(float v) {
#pragma unroll
  for (int m = 16; m >= 1; m >>= 1) v += __shfl_xor(v, m, 32);
  return v;
}

#define WMMA(A_, B_, C_) __builtin_amdgcn_wmma_f32_16x16x32_f16(false, A_, false, B_, (short)0, C_, false, false)

// ---------------------------------------------------------------------------
// Generic WMMA GEMM:  C[M,N] = A[M,K] * B^T   (B packed as [N][K], f16)
// Wave computes a 32x64 tile: 2 A frags x 4 B frags -> 8 WMMA per K-step,
// software-pipelined (next k-step fragments loaded before current WMMAs) so
// the s_wait_loadcnt overlaps the matrix ops instead of serializing them.
// Fragment layouts (A/B: this code's contiguous K order, not the ISA table; C/D per CDNA5 ISA 7.12.2):
//   A 16x32 f16 : lane l holds row (l&15), K = (l>>4)*16 .. +15  (contiguous)
//   B 32x16 f16 : lane l holds col (l&15), K = (l>>4)*16 .. +15  (contiguous)
//   C/D 16x16 f32: vgpr r, lane l -> (M = r + 8*(l>>4), N = (l&15))
// Epilogues (EPI):
//   0: outF = acc*sc[n] + sh[n]                      (lift + BN0 fused)
//   1: outH = f16(acc + sh[n])                       (edge transform / qk proj)
//   2: outH = f16(gelu(acc + sh[n]))                 (MLP up, exact erf GELU)
//   3: outF = acc + sh[n] + add0 + outF  (in place)  (MLP down + resi + edge)
//   4: d_out[nctv] = relu(acc*sc+sh + add0)          (down path + final add)
// ---------------------------------------------------------------------------
template<int EPI>
__global__ __launch_bounds__(256) void gemm_wmma(
    const _Float16* __restrict__ A, const _Float16* __restrict__ B,
    int Mrows, int Ncols, int Kdim,
    const float* __restrict__ sc, const float* __restrict__ sh,
    const float* __restrict__ add0, float* __restrict__ outF,
    _Float16* __restrict__ outH, float* __restrict__ outFin)
{
  const int wave   = (blockIdx.x * 256 + threadIdx.x) >> 5;
  const int lane   = threadIdx.x & 31;
  const int tilesN = Ncols >> 6;
  const int tM     = wave / tilesN;
  const int tN     = wave - tM * tilesN;
  const int m0     = tM << 5;            // 32 rows per wave
  const int n0     = tN << 6;            // 64 cols per wave
  const int lr     = lane & 15;          // A-row / B-col within 16-tile
  const int kh     = (lane >> 4) << 4;   // K half selector

  const _Float16* Ap0 = A + (size_t)(m0 + lr) * Kdim + kh;
  const _Float16* Ap1 = Ap0 + (size_t)16 * Kdim;
  const _Float16* Bp  = B + (size_t)(n0 + lr) * Kdim + kh;
  const size_t bs     = (size_t)16 * Kdim;

  v8f c00 = {}, c01 = {}, c02 = {}, c03 = {};
  v8f c10 = {}, c11 = {}, c12 = {}, c13 = {};

  // prologue: fragments for k0 = 0
  v16h a0 = *(const v16h*)(Ap0);
  v16h a1 = *(const v16h*)(Ap1);
  v16h b0 = *(const v16h*)(Bp);
  v16h b1 = *(const v16h*)(Bp + bs);
  v16h b2 = *(const v16h*)(Bp + 2 * bs);
  v16h b3 = *(const v16h*)(Bp + 3 * bs);

  for (int k0 = 32; k0 < Kdim; k0 += 32) {
    // issue next k-step loads first (overlap with the 8 WMMAs below)
    v16h a0n = *(const v16h*)(Ap0 + k0);
    v16h a1n = *(const v16h*)(Ap1 + k0);
    v16h b0n = *(const v16h*)(Bp + k0);
    v16h b1n = *(const v16h*)(Bp + bs + k0);
    v16h b2n = *(const v16h*)(Bp + 2 * bs + k0);
    v16h b3n = *(const v16h*)(Bp + 3 * bs + k0);
    __builtin_prefetch(Ap0 + k0 + 32, 0, 3);   // streaming A, keep near
    c00 = WMMA(a0, b0, c00);  c01 = WMMA(a0, b1, c01);
    c02 = WMMA(a0, b2, c02);  c03 = WMMA(a0, b3, c03);
    c10 = WMMA(a1, b0, c10);  c11 = WMMA(a1, b1, c11);
    c12 = WMMA(a1, b2, c12);  c13 = WMMA(a1, b3, c13);
    a0 = a0n; a1 = a1n; b0 = b0n; b1 = b1n; b2 = b2n; b3 = b3n;
  }
  // tail
  c00 = WMMA(a0, b0, c00);  c01 = WMMA(a0, b1, c01);
  c02 = WMMA(a0, b2, c02);  c03 = WMMA(a0, b3, c03);
  c10 = WMMA(a1, b0, c10);  c11 = WMMA(a1, b1, c11);
  c12 = WMMA(a1, b2, c12);  c13 = WMMA(a1, b3, c13);

  const int mrow0 = m0 + ((lane >> 4) << 3);
  v8f accs[2][4] = {{c00, c01, c02, c03}, {c10, c11, c12, c13}};
#pragma unroll
  for (int h = 0; h < 2; ++h) {
#pragma unroll
    for (int j = 0; j < 4; ++j) {
#pragma unroll
      for (int r = 0; r < 8; ++r) {
        float a  = accs[h][j][r];
        int   m  = mrow0 + h * 16 + r;
        int   nn = n0 + j * 16 + lr;
        size_t idx = (size_t)m * Ncols + nn;
        if (EPI == 0) {
          outF[idx] = a * sc[nn] + sh[nn];
        } else if (EPI == 1) {
          outH[idx] = (_Float16)(a + sh[nn]);
        } else if (EPI == 2) {
          float x = a + sh[nn];
          outH[idx] = (_Float16)(0.5f * x * (1.0f + erff(x * 0.70710678118654752f)));
        } else if (EPI == 3) {
          outF[idx] = a + sh[nn] + add0[idx] + outF[idx];
        } else {
          float v = a * sc[nn] + sh[nn] + add0[idx];
          v = fmaxf(v, 0.0f);
          int n = m / (TT * VV); int r2 = m - n * (TT * VV);
          int t = r2 / VV;       int vv = r2 - t * VV;
          outFin[(((size_t)n * CC + nn) * TT + t) * VV + vv] = v;
        }
      }
    }
  }
}

// ---------------------------------------------------------------------------
// LayerNorm over channels (per position row of [M,256]), f32 -> f16*g+b
// one wave per row, 8 channels per lane, shfl_xor reductions
// ---------------------------------------------------------------------------
__global__ __launch_bounds__(256) void ln_kernel(
    const float* __restrict__ x, const float* __restrict__ g,
    const float* __restrict__ b, _Float16* __restrict__ out)
{
  int lane = threadIdx.x & 31;
  int row  = blockIdx.x * 8 + (threadIdx.x >> 5);
  const float* xr = x + (size_t)row * CC;
  float v[8]; float s = 0.f;
#pragma unroll
  for (int i = 0; i < 8; ++i) { v[i] = xr[i * 32 + lane]; s += v[i]; }
  s = wsum32(s);
  float mu = s * (1.f / 256.f);
  float s2 = 0.f;
#pragma unroll
  for (int i = 0; i < 8; ++i) { float d = v[i] - mu; s2 += d * d; }
  s2 = wsum32(s2);
  float rstd = rsqrtf(s2 * (1.f / 256.f) + 1e-5f);
  _Float16* orow = out + (size_t)row * CC;
#pragma unroll
  for (int i = 0; i < 8; ++i) {
    int c = i * 32 + lane;
    orow[c] = (_Float16)((v[i] - mu) * rstd * g[c] + b[c]);
  }
}

// ---------------------------------------------------------------------------
// Edge-conv rank-6 bottleneck: per row, t6[e]=tanh((ef^T xt)/256)*a_e,
// edge = ew * ef @ t6.   One wave per row.
// ---------------------------------------------------------------------------
__global__ __launch_bounds__(256) void edge_kernel(
    const _Float16* __restrict__ xt, const float* __restrict__ ef,
    const float* __restrict__ ae, const float* __restrict__ ew,
    float* __restrict__ edge)
{
  int lane = threadIdx.x & 31;
  int row  = blockIdx.x * 8 + (threadIdx.x >> 5);
  const _Float16* xr = xt + (size_t)row * CC;
  float xv[8];
#pragma unroll
  for (int i = 0; i < 8; ++i) xv[i] = (float)xr[i * 32 + lane];
  float t6[EE];
#pragma unroll
  for (int e = 0; e < EE; ++e) {
    float s = 0.f;
#pragma unroll
    for (int i = 0; i < 8; ++i) { int c = i * 32 + lane; s += xv[i] * ef[c * EE + e]; }
    s = wsum32(s);
    t6[e] = tanhf(s * (1.f / 256.f)) * ae[e];
  }
  float w = ew[0];
  float* er = edge + (size_t)row * CC;
#pragma unroll
  for (int i = 0; i < 8; ++i) {
    int c = i * 32 + lane; float s = 0.f;
#pragma unroll
    for (int e = 0; e < EE; ++e) s += ef[c * EE + e] * t6[e];
    er[c] = s * w;
  }
}

// ---------------------------------------------------------------------------
// attn[n,k,u,w] = tanh( sum_{c,t} q[n,k,c,t,u]*k[n,k,c,t,w] / 2048 ) * alpha[k]
// qk stored [M,512] f16 (q channels k*32+c, k channels 256+k*32+c)
// ---------------------------------------------------------------------------
__global__ __launch_bounds__(256) void attn_scores_kernel(
    const _Float16* __restrict__ qk, const float* __restrict__ alphas,
    float* __restrict__ attnS)
{
  int o = blockIdx.x * 256 + threadIdx.x;
  if (o >= NB * KH * VV * VV) return;
  int w = o % VV; int r = o / VV;
  int u = r % VV; r /= VV;
  int k = r % KH; int n = r / KH;
  float s = 0.f;
  for (int t = 0; t < TT; ++t) {
    const _Float16* q  = qk + (size_t)((n * TT + t) * VV + u) * 512 + k * CIH;
    const _Float16* kk = qk + (size_t)((n * TT + t) * VV + w) * 512 + 256 + k * CIH;
#pragma unroll
    for (int c = 0; c < CIH; ++c) s += (float)q[c] * (float)kk[c];
  }
  attnS[o] = tanhf(s * (1.f / 2048.f)) * alphas[k];
}

// ---------------------------------------------------------------------------
// x[p,d] += sum_v xn[n,t,v,d] * attnS[n,k(d),v,w(p)]   (in-place residual)
// ---------------------------------------------------------------------------
__global__ __launch_bounds__(256) void attn_apply_kernel(
    const _Float16* __restrict__ xn, const float* __restrict__ attnS,
    float* __restrict__ x)
{
  size_t idx = (size_t)blockIdx.x * 256 + threadIdx.x;
  int d = (int)(idx & 255); size_t p = idx >> 8;
  int n = (int)(p / (TT * VV)); int r = (int)(p % (TT * VV));
  int t = r / VV; int w = r - t * VV;
  int k = d >> 5;
  const _Float16* xb = xn + ((size_t)(n * TT + t) * VV) * CC + d;
  const float*    ab = attnS + ((size_t)(n * KH + k) * VV) * VV + w;
  float s = 0.f;
#pragma unroll
  for (int v = 0; v < VV; ++v) s += (float)xb[(size_t)v * CC] * ab[v * VV];
  x[idx] += s;
}

// ---------------------------------------------------------------------------
// y[p,d] = BN( sum_v x[n,t,v,d] * att0[d,v,w(p)] )
// ---------------------------------------------------------------------------
__global__ __launch_bounds__(256) void att0_apply_kernel(
    const float* __restrict__ x, const float* __restrict__ att0,
    const float* __restrict__ sc, const float* __restrict__ sh,
    float* __restrict__ y)
{
  size_t idx = (size_t)blockIdx.x * 256 + threadIdx.x;
  int d = (int)(idx & 255); size_t p = idx >> 8;
  int n = (int)(p / (TT * VV)); int r = (int)(p % (TT * VV));
  int t = r / VV; int w = r - t * VV;
  const float* xb = x + ((size_t)(n * TT + t) * VV) * CC + d;
  const float* ab = att0 + (size_t)d * VV * VV + w;
  float s = 0.f;
#pragma unroll
  for (int v = 0; v < VV; ++v) s += xb[(size_t)v * CC] * ab[v * VV];
  y[idx] = s * sc[d] + sh[d];
}

// ---------------------------------------------------------------------------
// Fold BN stats (+ biases) into per-channel scale/shift
// ---------------------------------------------------------------------------
__global__ void prep_chan_kernel(
    const float* Lb, const float* bn0g, const float* bn0b, const float* bn0m, const float* bn0v,
    const float* bng, const float* bnb, const float* bnm, const float* bnv,
    const float* db, const float* dbng, const float* dbnb, const float* dbnm, const float* dbnv,
    float* sc0, float* sh0, float* scF, float* shF, float* scD, float* shD)
{
  int d = threadIdx.x;
  if (d < 256) {
    float s = bn0g[d] * rsqrtf(bn0v[d] + 1e-5f);
    sc0[d] = s; sh0[d] = (Lb[d] - bn0m[d]) * s + bn0b[d];
    s = bng[d] * rsqrtf(bnv[d] + 1e-5f);
    scF[d] = s; shF[d] = bnb[d] - bnm[d] * s;
    s = dbng[d] * rsqrtf(dbnv[d] + 1e-5f);
    scD[d] = s; shD[d] = (db[d] - dbnm[d]) * s + dbnb[d];
  }
}

// ---------------------------------------------------------------------------
// Packing kernels
// ---------------------------------------------------------------------------
__global__ __launch_bounds__(256) void pack_x0_kernel(
    const float* __restrict__ x0, _Float16* __restrict__ a0)
{
  size_t idx = (size_t)blockIdx.x * 256 + threadIdx.x;  // M*64 exact
  int c = (int)(idx & 63); size_t p = idx >> 6;
  int n = (int)(p / (TT * VV)); int r = (int)(p % (TT * VV));
  int t = r / VV; int v = r - t * VV;
  a0[idx] = (_Float16)x0[(((size_t)n * CINN + c) * TT + t) * VV + v];
}

__global__ __launch_bounds__(256) void convert_kernel(
    const float* __restrict__ s, _Float16* __restrict__ d, int nElem)
{
  int i = blockIdx.x * 256 + threadIdx.x;
  if (i < nElem) d[i] = (_Float16)s[i];
}

__global__ __launch_bounds__(256) void transpose_kernel(
    const float* __restrict__ s, _Float16* __restrict__ d, int R, int Ccols)
{
  int i = blockIdx.x * 256 + threadIdx.x;
  if (i >= R * Ccols) return;
  int r = i / Ccols, c = i - r * Ccols;
  d[(size_t)c * R + r] = (_Float16)s[i];
}

// ---------------------------------------------------------------------------
// Host side
// ---------------------------------------------------------------------------
extern "C" void kernel_launch(void* const* d_in, const int* in_sizes, int n_in,
                              void* d_out, int out_size, void* d_ws, size_t ws_size,
                              hipStream_t stream) {
  (void)in_sizes; (void)n_in; (void)out_size; (void)ws_size;
  const float* x0      = (const float*)d_in[0];
  const float* Lw      = (const float*)d_in[1];
  const float* Lb      = (const float*)d_in[2];
  const float* bn0_g   = (const float*)d_in[3];
  const float* bn0_b   = (const float*)d_in[4];
  const float* bn0_m   = (const float*)d_in[5];
  const float* bn0_v   = (const float*)d_in[6];
  const float* ec_ef   = (const float*)d_in[7];
  const float* ec_g    = (const float*)d_in[8];
  const float* ec_b    = (const float*)d_in[9];
  const float* ec_tw   = (const float*)d_in[10];
  const float* ec_tb   = (const float*)d_in[11];
  const float* ec_al   = (const float*)d_in[12];
  const float* edge_w  = (const float*)d_in[13];
  const float* n1_g    = (const float*)d_in[14];
  const float* n1_b    = (const float*)d_in[15];
  const float* qk_w    = (const float*)d_in[16];
  const float* qk_b    = (const float*)d_in[17];
  const float* alphas  = (const float*)d_in[18];
  const float* n2_g    = (const float*)d_in[19];
  const float* n2_b    = (const float*)d_in[20];
  const float* mlp_w1  = (const float*)d_in[21];
  const float* mlp_b1  = (const float*)d_in[22];
  const float* mlp_w2  = (const float*)d_in[23];
  const float* mlp_b2  = (const float*)d_in[24];
  const float* att0    = (const float*)d_in[25];
  const float* bn_g    = (const float*)d_in[26];
  const float* bn_b    = (const float*)d_in[27];
  const float* bn_m    = (const float*)d_in[28];
  const float* bn_v    = (const float*)d_in[29];
  const float* dw      = (const float*)d_in[30];
  const float* db      = (const float*)d_in[31];
  const float* dbn_g   = (const float*)d_in[32];
  const float* dbn_b   = (const float*)d_in[33];
  const float* dbn_m   = (const float*)d_in[34];
  const float* dbn_v   = (const float*)d_in[35];

  char* ws = (char*)d_ws;
  size_t off = 0;
  auto alloc = [&](size_t bytes) -> char* {
    char* p = ws + off;
    off = (off + bytes + 255) & ~(size_t)255;
    return p;
  };

  _Float16* a0    = (_Float16*)alloc((size_t)MM * 64 * 2);
  _Float16* wL    = (_Float16*)alloc((size_t)256 * 64 * 2);
  _Float16* wtw   = (_Float16*)alloc((size_t)256 * 256 * 2);
  _Float16* wqk   = (_Float16*)alloc((size_t)512 * 256 * 2);
  _Float16* w1t   = (_Float16*)alloc((size_t)1024 * 256 * 2);
  _Float16* w2t   = (_Float16*)alloc((size_t)256 * 1024 * 2);
  _Float16* wdw   = (_Float16*)alloc((size_t)256 * 64 * 2);
  float*    sc0   = (float*)alloc(256 * 4);
  float*    sh0   = (float*)alloc(256 * 4);
  float*    scF   = (float*)alloc(256 * 4);
  float*    shF   = (float*)alloc(256 * 4);
  float*    scD   = (float*)alloc(256 * 4);
  float*    shD   = (float*)alloc(256 * 4);
  float*    x     = (float*)alloc((size_t)MM * CC * 4);
  _Float16* xn16  = (_Float16*)alloc((size_t)MM * CC * 2);
  _Float16* xt16  = (_Float16*)alloc((size_t)MM * CC * 2);
  float*    edge  = (float*)alloc((size_t)MM * CC * 4);   // later reused as y
  _Float16* qk16  = (_Float16*)alloc((size_t)MM * 512 * 2);
  float*    attnS = (float*)alloc((size_t)NB * KH * VV * VV * 4);
  _Float16* h16   = (_Float16*)alloc((size_t)MM * HH * 2);

  auto gB = [](int Mr, int Nc) { return (Mr / 32) * (Nc / 64) / 8; };

  // --- preprocessing -------------------------------------------------------
  prep_chan_kernel<<<1, 256, 0, stream>>>(Lb, bn0_g, bn0_b, bn0_m, bn0_v,
                                          bn_g, bn_b, bn_m, bn_v,
                                          db, dbn_g, dbn_b, dbn_m, dbn_v,
                                          sc0, sh0, scF, shF, scD, shD);
  pack_x0_kernel<<<MM * 64 / 256, 256, 0, stream>>>(x0, a0);
  transpose_kernel<<<(64 * 256 + 255) / 256, 256, 0, stream>>>(Lw, wL, 64, 256);
  convert_kernel<<<(256 * 256 + 255) / 256, 256, 0, stream>>>(ec_tw, wtw, 256 * 256);
  convert_kernel<<<(512 * 256 + 255) / 256, 256, 0, stream>>>(qk_w, wqk, 512 * 256);
  transpose_kernel<<<(256 * 1024 + 255) / 256, 256, 0, stream>>>(mlp_w1, w1t, 256, 1024);
  transpose_kernel<<<(1024 * 256 + 255) / 256, 256, 0, stream>>>(mlp_w2, w2t, 1024, 256);
  convert_kernel<<<(256 * 64 + 255) / 256, 256, 0, stream>>>(dw, wdw, 256 * 64);

  // --- 1: x = BN0(x0 @ Lw + Lb)  [M,256] f32 -------------------------------
  gemm_wmma<0><<<gB(MM, 256), 256, 0, stream>>>(a0, wL, MM, 256, 64,
                                                sc0, sh0, nullptr, x, nullptr, nullptr);
  // --- 2: edge conv --------------------------------------------------------
  ln_kernel<<<MM / 8, 256, 0, stream>>>(x, ec_g, ec_b, xn16);
  gemm_wmma<1><<<gB(MM, 256), 256, 0, stream>>>(xn16, wtw, MM, 256, 256,
                                                nullptr, ec_tb, nullptr, nullptr, xt16, nullptr);
  edge_kernel<<<MM / 8, 256, 0, stream>>>(xt16, ec_ef, ec_al, edge_w, edge);
  // --- 3: attention --------------------------------------------------------
  ln_kernel<<<MM / 8, 256, 0, stream>>>(x, n1_g, n1_b, xn16);
  gemm_wmma<1><<<gB(MM, 512), 256, 0, stream>>>(xn16, wqk, MM, 512, 256,
                                                nullptr, qk_b, nullptr, nullptr, qk16, nullptr);
  attn_scores_kernel<<<(NB * KH * VV * VV + 255) / 256, 256, 0, stream>>>(qk16, alphas, attnS);
  attn_apply_kernel<<<MM, 256, 0, stream>>>(xn16, attnS, x);   // x += attn(xn)
  // --- 4: MLP --------------------------------------------------------------
  ln_kernel<<<MM / 8, 256, 0, stream>>>(x, n2_g, n2_b, xn16);
  gemm_wmma<2><<<gB(MM, 1024), 256, 0, stream>>>(xn16, w1t, MM, 1024, 256,
                                                 nullptr, mlp_b1, nullptr, nullptr, h16, nullptr);
  gemm_wmma<3><<<gB(MM, 256), 256, 0, stream>>>(h16, w2t, MM, 256, 1024,
                                                nullptr, mlp_b2, edge, x, nullptr, nullptr);
  // --- 5: att0 + final BN, y reuses edge buffer ----------------------------
  att0_apply_kernel<<<MM, 256, 0, stream>>>(x, att0, scF, shF, edge);
  // --- 6: down path GEMM + add + relu + NCTV scatter to d_out --------------
  gemm_wmma<4><<<gB(MM, 256), 256, 0, stream>>>(a0, wdw, MM, 256, 64,
                                                scD, shD, edge, nullptr, nullptr,
                                                (float*)d_out);
}
